// MLPBaseline_50483045598048
// MI455X (gfx1250) — compile-verified, matrix operand order checked by reading, not run on hardware
//
#include <hip/hip_runtime.h>

// ---------------- types ----------------
typedef __bf16 bf16_t;
typedef bf16_t v16bf __attribute__((ext_vector_type(16)));
typedef float  v8f   __attribute__((ext_vector_type(8)));
typedef unsigned int u32x4 __attribute__((ext_vector_type(4)));
typedef int i32x4 __attribute__((ext_vector_type(4)));
typedef int i32x8 __attribute__((ext_vector_type(8)));

union Frag { uint4 u4[2]; unsigned int u[8]; v16bf v; };

#define N_NODES 200000
#define FEAT    1301
#define KPAD    1312          // 41 * 32
#define KT      41
#define HID     256
#define BATCH   1024
#define SSITES  512

static __device__ __forceinline__ unsigned short f2bf(float f) {
    unsigned int u = __builtin_bit_cast(unsigned int, f);
    u += 0x7FFFu + ((u >> 16) & 1u);           // round-to-nearest-even
    return (unsigned short)(u >> 16);
}

// ---------------- prep: Wt[n][k] = bf16(W_in[k][n]), zero-padded K ----------------
__global__ void k_prep_w(const float* __restrict__ Win, unsigned short* __restrict__ Wt) {
    const int n = blockIdx.x;
    for (int k = threadIdx.x; k < KPAD; k += blockDim.x) {
        float v = (k < FEAT) ? Win[(long)k * HID + n] : 0.f;
        Wt[(long)n * KPAD + k] = f2bf(v);
    }
}

// ---------------- prep: wq = Wk @ q ; c0 = q . bk ----------------
__global__ void k_prep_q(const float* __restrict__ Wk, const float* __restrict__ q,
                         const float* __restrict__ bk, float* __restrict__ wq,
                         float* __restrict__ c0) {
    __shared__ float s_q[HID];
    __shared__ float s_red[HID];
    const int t = threadIdx.x;
    s_q[t]   = q[t];
    s_red[t] = q[t] * bk[t];
    __syncthreads();
    float acc = 0.f;
    const float* row = Wk + (long)t * HID;
    #pragma unroll 4
    for (int e = 0; e < HID; ++e) acc += row[e] * s_q[e];
    wq[t] = acc;
    for (int off = 128; off > 0; off >>= 1) {
        if (t < off) s_red[t] += s_red[t + off];
        __syncthreads();
    }
    if (t == 0) c0[0] = s_red[0];
}

// ---------------- TDM: DMA one 256x32 bf16 B-tile into LDS with hw row padding ----
// D# describes a 2D tensor at gsrc (row pitch KPAD bf16); tile 32(dim0) x 256(dim1).
// pad_interval=16 dwords (one 64B row), pad_amount=4 dwords -> LDS row pitch 20 dwords,
// matching the bank-conflict-free fragment layout used by the compute waves.
static __device__ __forceinline__ void tdm_load_b_tile(const unsigned short* gsrc,
                                                       unsigned int lds_off) {
    const unsigned long long ga = (unsigned long long)(size_t)gsrc;
    u32x4 g0;
    g0.x = 1u;                                          // count=1: valid user descriptor
    g0.y = lds_off;                                     // lds_addr (bytes)
    g0.z = (unsigned int)(ga & 0xFFFFFFFFu);            // global_addr[31:0]
    g0.w = (unsigned int)((ga >> 32) & 0x01FFFFFFu)     // global_addr[56:32]
         | (2u << 30);                                  // type = 2 ("image")
    i32x8 g1;
    g1[0] = (1 << 16)        // data_size = 2 bytes
          | (1 << 20)        // pad_enable
          | (3 << 22)        // pad_interval: 16 dwords between pads
          | (3 << 25);       // pad_amount: 4 dwords
    g1[1] = (32 << 16);      // tensor_dim0 = 32 (bits 79:48)
    g1[2] = (256 << 16);     // tensor_dim0 hi = 0 | tensor_dim1 = 256 (bits 111:80)
    g1[3] = (32 << 16);      // tensor_dim1 hi = 0 | tile_dim0 = 32 (bits 127:112)
    g1[4] = 256;             // tile_dim1 = 256, tile_dim2 = 0
    g1[5] = KPAD;            // tensor_dim0_stride = 1312 elements
    g1[6] = 0;               // stride hi / tensor_dim1_stride (unused for 2D)
    g1[7] = 0;
    const i32x4 z4 = {0, 0, 0, 0};
#if defined(__clang_major__) && (__clang_major__ >= 23)
    const i32x8 z8 = {0, 0, 0, 0, 0, 0, 0, 0};
    __builtin_amdgcn_tensor_load_to_lds(g0, g1, z4, z4, z8, 0);
#else
    __builtin_amdgcn_tensor_load_to_lds(g0, g1, z4, z4, 0);
#endif
}

static __device__ __forceinline__ void load_bfrag(Frag& f, const unsigned int* Blds,
                                                  int n, int l16, int sel) {
    const unsigned int* brow = &Blds[(n * 16 + l16) * 20];
    f.u4[0] = *(const uint4*)(brow + sel * 4);
    f.u4[1] = *(const uint4*)(brow + 8 + sel * 4);
}

// ---------------- GEMM1: h = elu(x @ W_in + b_in), bf16 WMMA, f32 accum ----------------
__global__ __launch_bounds__(256) void k_gemm_in(const float* __restrict__ x,
                                                 const unsigned short* __restrict__ Wt,
                                                 const float* __restrict__ b_in,
                                                 float* __restrict__ h) {
    __shared__ unsigned int Alds[128 * 20];   // 128 rows x 32 bf16 (16 dwords + 4 pad)
    __shared__ unsigned int Blds[256 * 20];   // 256 cols x 32 bf16 (16 dwords + 4 pad)

    const int tid  = threadIdx.x;
    const int wave = tid >> 5;
    const int lane = tid & 31;
    const int sel  = lane >> 4;       // half-wave K selector (ISA 16-bit A layout)
    const int l16  = lane & 15;
    const long rowBase = (long)blockIdx.x * 128;
    const int  m0 = wave * 16;

    v8f acc[16];
    #pragma unroll
    for (int n = 0; n < 16; ++n) acc[n] = (v8f){0.f,0.f,0.f,0.f,0.f,0.f,0.f,0.f};

    // A staging: thread t loads 16 consecutive f32 of one row-half, packs to 8 dwords
    const int ar = tid >> 1;
    const int ah = tid & 1;
    long agrow = rowBase + ar; if (agrow > (long)(N_NODES - 1)) agrow = N_NODES - 1;
    const float* xrow = x + agrow * (long)FEAT + ah * 16;
    unsigned int* adst = &Alds[ar * 20 + ah * 8];
    const unsigned int blds_off = (unsigned int)(size_t)(&Blds[0]);

    for (int kt = 0; kt < KT; ++kt) {
        const int k0 = kt * 32;
        __syncthreads();
        // ---- kick the Tensor Data Mover for the B tile (one wave issues) ----
        if (wave == 0) tdm_load_b_tile(Wt + k0, blds_off);
        // ---- stage A (f32 -> bf16 pairs) while the TDM runs ----
        {
            float f[16];
            if (k0 + 32 <= FEAT) {
                if (kt + 1 < KT) __builtin_prefetch(xrow + k0 + 32, 0, 0);
                #pragma unroll
                for (int j = 0; j < 16; ++j) f[j] = xrow[k0 + j];
            } else {
                #pragma unroll
                for (int j = 0; j < 16; ++j) {
                    const int kk = k0 + ah * 16 + j;
                    f[j] = (kk < FEAT) ? xrow[k0 + j] : 0.f;
                }
            }
            unsigned int p[8];
            #pragma unroll
            for (int i = 0; i < 8; ++i)
                p[i] = ((unsigned int)f2bf(f[2*i+1]) << 16) | (unsigned int)f2bf(f[2*i]);
            uint4 v0; v0.x=p[0]; v0.y=p[1]; v0.z=p[2]; v0.w=p[3];
            uint4 v1; v1.x=p[4]; v1.y=p[5]; v1.z=p[6]; v1.w=p[7];
            *(uint4*)(adst)     = v0;
            *(uint4*)(adst + 4) = v1;
        }
        if (wave == 0) __builtin_amdgcn_s_wait_tensorcnt(0);
        __syncthreads();

        // ---- fragments + 16 WMMAs, B-fragment double buffered ----
        Frag fa;
        const unsigned int* arow = &Alds[(m0 + l16) * 20];
        fa.u4[0] = *(const uint4*)(arow + sel * 4);
        fa.u4[1] = *(const uint4*)(arow + 8 + sel * 4);
        Frag fb[2];
        load_bfrag(fb[0], Blds, 0, l16, sel);
        #pragma unroll
        for (int n = 0; n < 16; ++n) {
            if (n + 1 < 16) load_bfrag(fb[(n + 1) & 1], Blds, n + 1, l16, sel);
            acc[n] = __builtin_amdgcn_wmma_f32_16x16x32_bf16(
                false, fa.v, false, fb[n & 1].v, (short)0, acc[n], false, false);
        }
    }

    // ---- epilogue: bias + ELU, guarded store ----
    #pragma unroll
    for (int n = 0; n < 16; ++n) {
        const int col = n * 16 + l16;
        const float bias = b_in[col];
        #pragma unroll
        for (int r = 0; r < 8; ++r) {
            const long grow = rowBase + m0 + r + 8 * sel;
            float v = acc[n][r] + bias;
            v = v > 0.f ? v : (__expf(v) - 1.f);
            if (grow < N_NODES) h[grow * HID + col] = v;
        }
    }
}

// ---------------- fused gather + scores + softmax + pool + MLP head ----------------
__global__ __launch_bounds__(256) void k_attn(const float* __restrict__ h,
                                              const int* __restrict__ sites,
                                              const int* __restrict__ lengths,
                                              const float* __restrict__ wq,
                                              const float* __restrict__ c0p,
                                              const float* __restrict__ W1,
                                              const float* __restrict__ b1,
                                              const float* __restrict__ W2,
                                              const float* __restrict__ b2,
                                              float* __restrict__ out) {
    __shared__ int   s_site[SSITES];
    __shared__ float s_sc[SSITES];
    __shared__ float s_wq[HID];
    __shared__ float s_pool[HID];
    __shared__ float s_red[HID];

    const int tid  = threadIdx.x;
    const int b    = blockIdx.x;
    const int wave = tid >> 5;
    const int lane = tid & 31;
    const int len  = lengths[b];
    const float c0 = c0p[0];
    const float scale = 0.0625f;   // 1/sqrt(256)

    s_wq[tid] = wq[tid];
    for (int s = tid; s < SSITES; s += 256) s_site[s] = sites[b * SSITES + s];
    __syncthreads();

    for (int s = wave; s < SSITES; s += 8) {
        const float* hr = h + (long)s_site[s] * HID + lane * 8;
        const float4 a0 = *(const float4*)hr;
        const float4 a1 = *(const float4*)(hr + 4);
        const float4 q0 = *(const float4*)&s_wq[lane * 8];
        const float4 q1 = *(const float4*)&s_wq[lane * 8 + 4];
        float d = a0.x*q0.x + a0.y*q0.y + a0.z*q0.z + a0.w*q0.w
                + a1.x*q1.x + a1.y*q1.y + a1.z*q1.z + a1.w*q1.w;
        #pragma unroll
        for (int off = 16; off > 0; off >>= 1) d += __shfl_xor(d, off);
        if (lane == 0) s_sc[s] = (s < len) ? (d + c0) * scale : -1e9f;
    }
    __syncthreads();

    float m = fmaxf(s_sc[tid], s_sc[tid + 256]);
    s_red[tid] = m; __syncthreads();
    for (int off = 128; off > 0; off >>= 1) {
        if (tid < off) s_red[tid] = fmaxf(s_red[tid], s_red[tid + off]);
        __syncthreads();
    }
    m = s_red[0]; __syncthreads();
    const float e0 = expf(s_sc[tid] - m);
    const float e1 = expf(s_sc[tid + 256] - m);
    s_sc[tid] = e0; s_sc[tid + 256] = e1;
    s_red[tid] = e0 + e1; __syncthreads();
    for (int off = 128; off > 0; off >>= 1) {
        if (tid < off) s_red[tid] += s_red[tid + off];
        __syncthreads();
    }
    const float inv = 1.f / s_red[0];
    __syncthreads();

    float acc = 0.f;
    for (int s = 0; s < len; ++s)
        acc += s_sc[s] * h[(long)s_site[s] * HID + tid];
    s_pool[tid] = acc * inv;
    __syncthreads();

    if (tid < 128) {
        float aj = b1[tid];
        #pragma unroll 4
        for (int d2 = 0; d2 < HID; ++d2) aj += s_pool[d2] * W1[d2 * 128 + tid];
        aj = fmaxf(aj, 0.f);
        s_red[tid] = aj * W2[tid];
    }
    __syncthreads();
    for (int off = 64; off > 0; off >>= 1) {
        if (tid < off) s_red[tid] += s_red[tid + off];
        __syncthreads();
    }
    if (tid == 0) out[b] = s_red[0] + b2[0];
}

// ---------------- launcher ----------------
extern "C" void kernel_launch(void* const* d_in, const int* in_sizes, int n_in,
                              void* d_out, int out_size, void* d_ws, size_t ws_size,
                              hipStream_t stream) {
    const float* x    = (const float*)d_in[0];
    // d_in[1] = edge_index : unused by the reference computation
    const int*   sites    = (const int*)d_in[2];
    const int*   lengths  = (const int*)d_in[3];
    const float* W_in = (const float*)d_in[4];
    const float* b_in = (const float*)d_in[5];
    const float* q    = (const float*)d_in[6];
    const float* Wk   = (const float*)d_in[7];
    const float* bk   = (const float*)d_in[8];
    const float* W1   = (const float*)d_in[9];
    const float* b1   = (const float*)d_in[10];
    const float* W2   = (const float*)d_in[11];
    const float* b2   = (const float*)d_in[12];

    char* ws = (char*)d_ws;
    const size_t H_BYTES  = (size_t)N_NODES * HID * sizeof(float);       // 204,800,000
    const size_t WT_BYTES = (size_t)HID * KPAD * sizeof(unsigned short); // 671,744
    float*          h  = (float*)ws;
    unsigned short* Wt = (unsigned short*)(ws + H_BYTES);
    float*          wq = (float*)(ws + H_BYTES + WT_BYTES);
    float*          c0 = wq + HID;
    float* out = (float*)d_out;

    k_prep_w<<<HID, 256, 0, stream>>>(W_in, Wt);
    k_prep_q<<<1, 256, 0, stream>>>(Wk, q, bk, wq, c0);
    const int mblocks = (N_NODES + 127) / 128;   // 1563
    k_gemm_in<<<mblocks, 256, 0, stream>>>(x, Wt, b_in, h);
    k_attn<<<BATCH, 256, 0, stream>>>(h, sites, lengths, wq, c0, W1, b1, W2, b2, out);
}
